// CausalSelfAttention_67723044323937
// MI455X (gfx1250) — compile-verified, matrix operand order checked by reading, not run on hardware
//
#include <hip/hip_runtime.h>
#include <hip/hip_bf16.h>

// ---------------------------------------------------------------------------
// Causal self-attention for MI455X (gfx1250), wave32 + WMMA f16->f32.
// Stages: cvt/transpose -> QKV GEMM -> flash attention (async-LDS staged)
//         -> proj GEMM.
// ---------------------------------------------------------------------------

typedef _Float16 v16h __attribute__((ext_vector_type(16)));
typedef _Float16 v8h  __attribute__((ext_vector_type(8)));
typedef float    v8f  __attribute__((ext_vector_type(8)));

#define WMMA_F16(a, b, c) \
  __builtin_amdgcn_wmma_f32_16x16x32_f16(false, (a), false, (b), (short)0, (c), false, false)

constexpr int BB = 4;       // batch
constexpr int TT = 2048;    // seq len
constexpr int CC = 1024;    // channels
constexpr int HH = 16;      // heads
constexpr int DD = 64;      // head dim
constexpr int MTOT = BB * TT;   // 8192 rows

// Load a 16x32 f16 fragment (A layout, or B layout from an N-major matrix).
// ISA 7.12.2: lane L holds row (L&15); halfs 0..7 at k = (L>>4)*8, halfs 8..15
// at k = (L>>4)*8 + 16. Two contiguous 16B loads per lane.
__device__ __forceinline__ v16h load_frag(const _Float16* tile, int ld) {
  const int lane = threadIdx.x & 31;
  const _Float16* p = tile + (size_t)(lane & 15) * ld + ((lane >> 4) << 3);
  v8h lo = *reinterpret_cast<const v8h*>(p);
  v8h hi = *reinterpret_cast<const v8h*>(p + 16);
  v16h f;
#pragma unroll
  for (int i = 0; i < 8; ++i) { f[i] = lo[i]; f[i + 8] = hi[i]; }
  return f;
}

__device__ __forceinline__ v8f zero8() {
  v8f z;
#pragma unroll
  for (int i = 0; i < 8; ++i) z[i] = 0.0f;
  return z;
}

// Async copy 16 bytes global -> LDS (per lane), tracked by ASYNCcnt.
// LDS offset = low 32 bits of the generic shared-memory address.
__device__ __forceinline__ void async_ld16(void* lds, const _Float16* g) {
  uint32_t off = (uint32_t)(uintptr_t)lds;
  asm volatile("global_load_async_to_lds_b128 %0, %1, off"
               :: "v"(off), "v"(g) : "memory");
}

__device__ __forceinline__ void wait_async0() {
  asm volatile("s_wait_asynccnt 0x0" ::: "memory");
}

// ------------------------- stage 0: conversions ----------------------------

__global__ __launch_bounds__(256) void cvt_f32_to_f16(const float* __restrict__ in,
                                                      _Float16* __restrict__ out, int n) {
  int i = blockIdx.x * 256 + threadIdx.x;
  if (i < n) out[i] = (_Float16)in[i];
}

// out[n][k] = (f16) in[k][n]   (in is [K][N] row-major)
__global__ __launch_bounds__(256) void transpose_cvt(const float* __restrict__ in,
                                                     _Float16* __restrict__ out,
                                                     int K, int N) {
  int i = blockIdx.x * 256 + threadIdx.x;
  if (i < K * N) {
    int nIdx = i / K;
    int kIdx = i - nIdx * K;
    out[i] = (_Float16)in[(size_t)kIdx * N + nIdx];
  }
}

// ------------------------- stage 1: QKV GEMM -------------------------------
// qkv[m][n] = xh[m][:] . WqkvT[n][:] + bqkv[n]; scatter into q/k (B,H,T,D)
// and v transposed (B,H,D,T), all f16.
__global__ __launch_bounds__(256) void gemm_qkv(const _Float16* __restrict__ A,   // [8192][1024]
                                                const _Float16* __restrict__ Bt,  // [3072][1024]
                                                const float* __restrict__ bias,   // [3072]
                                                _Float16* __restrict__ qb,
                                                _Float16* __restrict__ kb,
                                                _Float16* __restrict__ vtb) {
  const int K = CC;
  const int lane = threadIdx.x & 31;
  const int wave = threadIdx.x >> 5;
  const int wm = wave & 3, wn = wave >> 2;
  const int rowbase = blockIdx.x * 128 + wm * 32;
  const int colbase = blockIdx.y * 128 + wn * 64;

  v8f acc[2][4];
#pragma unroll
  for (int i = 0; i < 2; ++i)
#pragma unroll
    for (int j = 0; j < 4; ++j) acc[i][j] = zero8();

  for (int k0 = 0; k0 < K; k0 += 32) {
    v16h af[2], bf[4];
#pragma unroll
    for (int i = 0; i < 2; ++i)
      af[i] = load_frag(A + (size_t)(rowbase + i * 16) * K + k0, K);
#pragma unroll
    for (int j = 0; j < 4; ++j)
      bf[j] = load_frag(Bt + (size_t)(colbase + j * 16) * K + k0, K);
#pragma unroll
    for (int i = 0; i < 2; ++i)
#pragma unroll
      for (int j = 0; j < 4; ++j) acc[i][j] = WMMA_F16(af[i], bf[j], acc[i][j]);
  }

  const int hs = lane >> 4, ln = lane & 15;
#pragma unroll
  for (int j = 0; j < 4; ++j) {
    const int gn = colbase + j * 16 + ln;
    const float bv = bias[gn];
    const int sec = gn >> 10;            // 0=q 1=k 2=v (uniform across lanes per j)
    const int hh = (gn & 1023) >> 6;
    const int dd = gn & 63;
    if (sec == 2) {
#pragma unroll
      for (int i = 0; i < 2; ++i)
#pragma unroll
        for (int r = 0; r < 8; ++r) {
          const int gm = rowbase + i * 16 + r + 8 * hs;
          const int bidx = gm >> 11, t = gm & (TT - 1);
          vtb[(((size_t)bidx * HH + hh) * DD + dd) * TT + t] =
              (_Float16)(acc[i][j][r] + bv);
        }
    } else {
      _Float16* dst = (sec == 0) ? qb : kb;
#pragma unroll
      for (int i = 0; i < 2; ++i)
#pragma unroll
        for (int r = 0; r < 8; ++r) {
          const int gm = rowbase + i * 16 + r + 8 * hs;
          const int bidx = gm >> 11, t = gm & (TT - 1);
          dst[(((size_t)bidx * HH + hh) * TT + t) * DD + dd] =
              (_Float16)(acc[i][j][r] + bv);
        }
    }
  }
}

// ------------------------- stage 2: flash attention ------------------------
// One wave per 16-query tile; 8 waves/block share one (b,h) and cooperatively
// stage each 32-key K/V block into LDS with async global->LDS copies.
__global__ __launch_bounds__(256) void attn_fwd(const _Float16* __restrict__ q,   // [B,H,T,D]
                                                const _Float16* __restrict__ k,   // [B,H,T,D]
                                                const _Float16* __restrict__ vt,  // [B,H,D,T]
                                                _Float16* __restrict__ o) {       // [B,T,C]
  __shared__ _Float16 kbuf[32 * 64];     // [key][d]
  __shared__ _Float16 vbuf[64 * 32];     // [d][key]
  __shared__ _Float16 pbuf[8][16 * 32];  // per-wave P tile
  const int tid = threadIdx.x;
  const int lane = tid & 31;
  const int wave = tid >> 5;
  const int tile = blockIdx.x * 8 + wave;      // 8192 tiles; 128 tiles per (b,h)
  const int qt = tile & 127;
  const int bh = tile >> 7;                    // uniform across the block
  const int b = bh >> 4, h = bh & 15;
  const float scale = 0.125f;                  // D^-0.5

  const _Float16* qp = q + (size_t)bh * TT * DD + (size_t)qt * 16 * DD;
  const _Float16* kp = k + (size_t)bh * TT * DD;
  const _Float16* vp = vt + (size_t)bh * DD * TT;
  _Float16* pl = &pbuf[wave][0];

  const v16h qf0 = load_frag(qp, DD);
  const v16h qf1 = load_frag(qp + 32, DD);

  v8f oacc[4];
#pragma unroll
  for (int d = 0; d < 4; ++d) oacc[d] = zero8();
  float mrow[8], lrow[8];
#pragma unroll
  for (int r = 0; r < 8; ++r) { mrow[r] = -1e30f; lrow[r] = 0.0f; }

  const int hs = lane >> 4, ln = lane & 15;
  const int kend_w = qt * 16 + 16;                         // this wave's key limit
  const int qt_base = (blockIdx.x & 15) * 8;               // 16 blocks per (b,h)
  const int kend_max = (qt_base + 7) * 16 + 16;            // block-uniform bound

  // cooperative staging assignments (256 threads)
  const int krow = tid >> 3, kseg = tid & 7;   // K: 32 rows x 8 x 16B
  const int vrow = tid >> 2, vseg = tid & 3;   // V: 64 rows x 4 x 16B

  for (int kb = 0; kb < kend_max; kb += 32) {
    // stage K[kb..kb+31][0..63] and Vt[0..63][kb..kb+31] into LDS
    async_ld16(&kbuf[krow * 64 + kseg * 8], kp + (size_t)(kb + krow) * DD + kseg * 8);
    async_ld16(&vbuf[vrow * 32 + vseg * 8], vp + (size_t)vrow * TT + kb + vseg * 8);
    wait_async0();
    __syncthreads();

    if (kb < kend_w) {   // wave-uniform; EXEC stays all-ones inside
      // S = Q . K^T for 16 queries x 32 keys (from LDS)
      v16h b00 = load_frag(kbuf, 64);
      v16h b01 = load_frag(kbuf + 32, 64);
      v16h b10 = load_frag(kbuf + 16 * 64, 64);
      v16h b11 = load_frag(kbuf + 16 * 64 + 32, 64);
      v8f s0 = zero8(), s1 = zero8();
      s0 = WMMA_F16(qf0, b00, s0);
      s0 = WMMA_F16(qf1, b01, s0);
      s1 = WMMA_F16(qf0, b10, s1);
      s1 = WMMA_F16(qf1, b11, s1);

      // online softmax per row; write P tile (f16) to wave-private LDS
#pragma unroll
      for (int r = 0; r < 8; ++r) {
        const int rowg = qt * 16 + r + 8 * hs;
        const int c0 = kb + ln, c1 = c0 + 16;
        float x0 = (c0 <= rowg) ? s0[r] * scale : -1e30f;
        float x1 = (c1 <= rowg) ? s1[r] * scale : -1e30f;
        float mx = fmaxf(x0, x1);
#pragma unroll
        for (int d = 1; d < 16; d <<= 1) mx = fmaxf(mx, __shfl_xor(mx, d, 32));
        const float mnew = fmaxf(mrow[r], mx);
        const float corr = __expf(mrow[r] - mnew);
        const float p0 = __expf(x0 - mnew);
        const float p1 = __expf(x1 - mnew);
        float ps = p0 + p1;
#pragma unroll
        for (int d = 1; d < 16; d <<= 1) ps += __shfl_xor(ps, d, 32);
        lrow[r] = lrow[r] * corr + ps;
        mrow[r] = mnew;
#pragma unroll
        for (int dch = 0; dch < 4; ++dch) oacc[dch][r] *= corr;
        const int mloc = r + 8 * hs;
        pl[mloc * 32 + ln] = (_Float16)p0;
        pl[mloc * 32 + ln + 16] = (_Float16)p1;
      }
      // wave-local LDS RAW: wait for ds stores, block compiler reordering
      asm volatile("s_wait_dscnt 0" ::: "memory");
      const v16h pf = load_frag(pl, 32);

      // O += P . V (Vt tile in LDS: [d][key], contiguous along keys)
#pragma unroll
      for (int dch = 0; dch < 4; ++dch) {
        v16h vf = load_frag(vbuf + dch * 16 * 32, 32);
        oacc[dch] = WMMA_F16(pf, vf, oacc[dch]);
      }
    }
    __syncthreads();   // protect LDS tiles before next staging round
  }

  // finalize: divide by row sums, write f16 [B,T,C]
#pragma unroll
  for (int r = 0; r < 8; ++r) {
    const float inv = 1.0f / lrow[r];
    const size_t gm = (size_t)b * TT + qt * 16 + r + 8 * hs;
#pragma unroll
    for (int dch = 0; dch < 4; ++dch) {
      o[gm * CC + h * 64 + dch * 16 + ln] = (_Float16)(oacc[dch][r] * inv);
    }
  }
}

// ------------------------- stage 3: projection GEMM ------------------------
__global__ __launch_bounds__(256) void gemm_proj(const _Float16* __restrict__ A,   // [8192][1024]
                                                 const _Float16* __restrict__ Bt,  // [1024][1024]
                                                 const float* __restrict__ bias,   // [1024]
                                                 float* __restrict__ out) {        // [8192][1024]
  const int K = CC, N = CC;
  const int lane = threadIdx.x & 31;
  const int wave = threadIdx.x >> 5;
  const int wm = wave & 3, wn = wave >> 2;
  const int rowbase = blockIdx.x * 128 + wm * 32;
  const int colbase = blockIdx.y * 128 + wn * 64;

  v8f acc[2][4];
#pragma unroll
  for (int i = 0; i < 2; ++i)
#pragma unroll
    for (int j = 0; j < 4; ++j) acc[i][j] = zero8();

  for (int k0 = 0; k0 < K; k0 += 32) {
    v16h af[2], bf[4];
#pragma unroll
    for (int i = 0; i < 2; ++i)
      af[i] = load_frag(A + (size_t)(rowbase + i * 16) * K + k0, K);
#pragma unroll
    for (int j = 0; j < 4; ++j)
      bf[j] = load_frag(Bt + (size_t)(colbase + j * 16) * K + k0, K);
#pragma unroll
    for (int i = 0; i < 2; ++i)
#pragma unroll
      for (int j = 0; j < 4; ++j) acc[i][j] = WMMA_F16(af[i], bf[j], acc[i][j]);
  }

  const int hs = lane >> 4, ln = lane & 15;
#pragma unroll
  for (int j = 0; j < 4; ++j) {
    const int gn = colbase + j * 16 + ln;
    const float bv = bias[gn];
#pragma unroll
    for (int i = 0; i < 2; ++i) {
#pragma unroll
      for (int r = 0; r < 8; ++r) {
        const int gm = rowbase + i * 16 + r + 8 * hs;
        out[(size_t)gm * N + gn] = acc[i][j][r] + bv;
      }
    }
  }
}

// ---------------------------------------------------------------------------

extern "C" void kernel_launch(void* const* d_in, const int* in_sizes, int n_in,
                              void* d_out, int out_size, void* d_ws, size_t ws_size,
                              hipStream_t stream) {
  const float* x     = (const float*)d_in[0];  // [4,2048,1024]
  const float* Wqkv  = (const float*)d_in[1];  // [1024,3072]
  const float* bqkv  = (const float*)d_in[2];  // [3072]
  const float* Wproj = (const float*)d_in[3];  // [1024,1024]
  const float* bproj = (const float*)d_in[4];  // [1024]
  float* out = (float*)d_out;

  // workspace carve-up (f16 elements); total ~88 MB
  _Float16* ws = (_Float16*)d_ws;
  _Float16* xh     = ws;                              // 8192*1024
  _Float16* wqkvt  = xh    + (size_t)MTOT * CC;       // 3072*1024
  _Float16* wprojt = wqkvt + (size_t)3 * CC * CC;     // 1024*1024
  _Float16* qb     = wprojt + (size_t)CC * CC;        // B*H*T*D
  _Float16* kb     = qb    + (size_t)BB * HH * TT * DD;
  _Float16* vtb    = kb    + (size_t)BB * HH * TT * DD;
  _Float16* attnh  = vtb   + (size_t)BB * HH * TT * DD; // 8192*1024

  // stage 0: conversions
  {
    int n = MTOT * CC;
    cvt_f32_to_f16<<<(n + 255) / 256, 256, 0, stream>>>(x, xh, n);
    int nq = CC * 3 * CC;
    transpose_cvt<<<(nq + 255) / 256, 256, 0, stream>>>(Wqkv, wqkvt, CC, 3 * CC);
    int np = CC * CC;
    transpose_cvt<<<(np + 255) / 256, 256, 0, stream>>>(Wproj, wprojt, CC, CC);
  }

  // stage 1: QKV GEMM (M=8192, N=3072, K=1024)
  gemm_qkv<<<dim3(MTOT / 128, (3 * CC) / 128), 256, 0, stream>>>(
      xh, wqkvt, bqkv, qb, kb, vtb);

  // stage 2: flash attention (8192 query tiles, 8 waves/block)
  attn_fwd<<<(BB * HH * (TT / 16)) / 8, 256, 0, stream>>>(qb, kb, vtb, attnh);

  // stage 3: projection GEMM (M=8192, N=1024, K=1024) -> f32 out
  gemm_proj<<<dim3(MTOT / 128, CC / 128), 256, 0, stream>>>(
      attnh, wprojt, bproj, out);
}
